// PatchInpainting_74363063763313
// MI455X (gfx1250) — matrix operand order repaired and checked by reading, not compiled, not run on hardware
//
#include <hip/hip_runtime.h>
#include <hip/hip_bf16.h>
#include <math.h>

// ---------------------------------------------------------------------------
// Patch-inpainting attention pipeline for MI455X (gfx1250, wave32).
// All big GEMMs use v_wmma_f32_16x16x32_bf16 (bf16 in, f32 accumulate).
// Roofline: ~107 GFLOP vs ~300MB traffic @ 23.3 TB/s -> compute-bound,
// so precision is dropped to bf16 for the matrix cores; softmax stays f32.
// A-tiles are staged with GLOBAL_LOAD_ASYNC_TO_LDS_B128 (ASYNCcnt DMA);
// B-tiles are transposed during staging via a two-phase register pass
// (loads pipelined, then ds_store_b16 scatter).
// ---------------------------------------------------------------------------

#define HW    512
#define GG    32
#define NP    1024      // 32*32 patches
#define PVD   768       // 3*16*16
#define FD    128
#define PTD   896       // PVD + FD
#define DQK   2048
#define BATCH 8

typedef __attribute__((ext_vector_type(16))) __bf16 bf16x16;
typedef __attribute__((ext_vector_type(8)))  float  f32x8;
typedef __attribute__((ext_vector_type(4)))  int    i32x4;

union FragAB { bf16x16 bf; uint4 u4[2]; };

__device__ __forceinline__ unsigned short f32_to_bf16_rne(float f) {
  unsigned int u = __float_as_uint(f);
  u += 0x7FFFu + ((u >> 16) & 1u);          // round-to-nearest-even
  return (unsigned short)(u >> 16);
}

// ---------------------------------------------------------------------------
// 1) 3x3 stem conv (pad 1), writes coarse_raw (also a kernel output)
// ---------------------------------------------------------------------------
__global__ void stem_conv_kernel(const float* __restrict__ img,
                                 const float* __restrict__ w,
                                 float* __restrict__ coarse) {
  long long idx = (long long)blockIdx.x * blockDim.x + threadIdx.x;
  const long long total = (long long)BATCH * 3 * HW * HW;
  if (idx >= total) return;
  int x = (int)(idx & (HW - 1));
  int y = (int)((idx >> 9) & (HW - 1));
  int bo = (int)(idx >> 18);
  int o = bo % 3, b = bo / 3;
  float s = 0.f;
  #pragma unroll
  for (int i = 0; i < 3; ++i) {
    #pragma unroll
    for (int ky = 0; ky < 3; ++ky) {
      int yy = y + ky - 1;
      if (yy < 0 || yy >= HW) continue;
      #pragma unroll
      for (int kx = 0; kx < 3; ++kx) {
        int xx = x + kx - 1;
        if (xx < 0 || xx >= HW) continue;
        s += img[((size_t)(b * 3 + i) * HW + yy) * HW + xx] *
             w[((o * 3 + i) * 3 + ky) * 3 + kx];
      }
    }
  }
  coarse[idx] = s;
}

// ---------------------------------------------------------------------------
// 2) f32 -> bf16 elementwise convert (for the QK weight matrices)
// ---------------------------------------------------------------------------
__global__ void f32_to_bf16_kernel(const float* __restrict__ in,
                                   unsigned short* __restrict__ out,
                                   long long n) {
  long long i = (long long)blockIdx.x * blockDim.x + threadIdx.x;
  if (i < n) out[i] = f32_to_bf16_rne(in[i]);
}

// ---------------------------------------------------------------------------
// 3) Token assembly: tok[b][p][d] = unfold(coarse)|bilinear(feat) + pos, bf16
//    bilinear 128->32 with scale 4 reduces exactly to a 2x2 average at
//    (4i+1, 4j+1) since the fractional weights are all 0.5.
// ---------------------------------------------------------------------------
__global__ void build_tokens_kernel(const float* __restrict__ coarse,
                                    const float* __restrict__ feat,
                                    const float* __restrict__ pos,
                                    unsigned short* __restrict__ tok) {
  long long idx = (long long)blockIdx.x * blockDim.x + threadIdx.x;
  const long long total = (long long)BATCH * NP * PTD;
  if (idx >= total) return;
  int d = (int)(idx % PTD);
  int p = (int)((idx / PTD) & (NP - 1));
  int b = (int)(idx / ((long long)PTD * NP));
  int py = p >> 5, px = p & 31;
  float v;
  if (d < PVD) {
    int c = d >> 8, rem = d & 255, ky = rem >> 4, kx = rem & 15;
    v = coarse[((size_t)(b * 3 + c) * HW + py * 16 + ky) * HW + px * 16 + kx];
  } else {
    int c = d - PVD;
    const float* f = feat + (size_t)(b * FD + c) * 128 * 128;
    int y0 = 4 * py + 1, x0 = 4 * px + 1;
    v = 0.25f * (f[y0 * 128 + x0] + f[y0 * 128 + x0 + 1] +
                 f[(y0 + 1) * 128 + x0] + f[(y0 + 1) * 128 + x0 + 1]);
  }
  v += pos[(size_t)d * NP + p];
  tok[idx] = f32_to_bf16_rne(v);
}

// ---------------------------------------------------------------------------
// 4) Value matrix, stored transposed: vt[b][v][p] = unfold(image)[b][v][p]
//    (value-major so the attn@V GEMM is the same A @ B^T form, K-contiguous)
// ---------------------------------------------------------------------------
__global__ void build_values_kernel(const float* __restrict__ img,
                                    unsigned short* __restrict__ vt) {
  long long idx = (long long)blockIdx.x * blockDim.x + threadIdx.x;
  const long long total = (long long)BATCH * PVD * NP;
  if (idx >= total) return;
  int p = (int)(idx & (NP - 1));
  int v = (int)((idx >> 10) % PVD);
  int b = (int)(idx / ((long long)PVD * NP));
  int py = p >> 5, px = p & 31;
  int c = v >> 8, rem = v & 255, ky = rem >> 4, kx = rem & 15;
  float val = img[((size_t)(b * 3 + c) * HW + py * 16 + ky) * HW + px * 16 + kx];
  vt[idx] = f32_to_bf16_rne(val);
}

// ---------------------------------------------------------------------------
// 5) Per-patch mask flag: any mask pixel in the 16x16 patch > 0
// ---------------------------------------------------------------------------
__global__ void mask_flags_kernel(const float* __restrict__ mask,
                                  float* __restrict__ mflag) {
  int idx = blockIdx.x * blockDim.x + threadIdx.x;
  if (idx >= BATCH * NP) return;
  int p = idx & (NP - 1);
  int b = idx >> 10;
  int py = p >> 5, px = p & 31;
  float mx = 0.f;
  for (int ky = 0; ky < 16; ++ky)
    for (int kx = 0; kx < 16; ++kx)
      mx = fmaxf(mx, mask[((size_t)b * HW + py * 16 + ky) * HW + px * 16 + kx]);
  mflag[idx] = (mx > 0.f) ? 1.f : 0.f;
}

// ---------------------------------------------------------------------------
// 6) bf16 WMMA GEMM:  C[M,N] = alpha * A[M,K] @ B[N,K]^T   (row-major, K-contig)
//    Block tile 128x128, 8 waves (2x4), each wave 4x2 tiles of 16x16.
//    A staged row-major in LDS (async DMA when available); B staged transposed
//    [k][n]; each lane's B fragment (lane -> n, element s -> k) is sixteen strided
//    16-bit LDS reads.
// ---------------------------------------------------------------------------
__global__ void __launch_bounds__(256)
gemm_bf16_wmma(const unsigned short* __restrict__ Ag,
               const unsigned short* __restrict__ Bg,
               void* __restrict__ Cg,
               int M, int N, int Kd,
               long long sA, long long sB, long long sC,
               float alpha, int outBf16) {
  __shared__ __align__(16) unsigned short As[128][32];   // [m][k]
  __shared__ __align__(16) unsigned short Bs[32][128];   // [k][n] (transposed)

  const unsigned short* A = Ag + (size_t)blockIdx.z * sA;
  const unsigned short* B = Bg + (size_t)blockIdx.z * sB;

  const int tid   = threadIdx.x;
  const int lane  = tid & 31;
  const int wave  = tid >> 5;
  const int wm    = wave >> 2;   // 0..1  -> 64 rows
  const int wn    = wave & 3;    // 0..3  -> 32 cols
  const int half  = lane >> 4;
  const int mlane = lane & 15;

  const int m0 = blockIdx.y * 128;
  const int n0 = blockIdx.x * 128;

  f32x8 acc[4][2];
  #pragma unroll
  for (int i = 0; i < 4; ++i)
    #pragma unroll
    for (int j = 0; j < 2; ++j)
      #pragma unroll
      for (int r = 0; r < 8; ++r) acc[i][j][r] = 0.f;

  const int nk = Kd >> 5;
  for (int kt = 0; kt < nk; ++kt) {
    const int k0 = kt << 5;

    // ---- Stage A tile: 128 rows x 32 bf16 (2 x 16B chunks per thread) -----
#if __has_builtin(__builtin_amdgcn_global_load_async_to_lds_b128) && \
    __has_builtin(__builtin_amdgcn_s_wait_asynccnt)
    // CDNA5 async DMA global->LDS: per-lane global src + LDS dst, ASYNCcnt.
    #pragma unroll
    for (int t = 0; t < 2; ++t) {
      int idx = tid + t * 256;           // 0..511
      int r = idx >> 2, ch = idx & 3;    // row, 16-byte chunk
      const unsigned short* g = A + (size_t)(m0 + r) * Kd + k0 + (ch << 3);
      __builtin_amdgcn_global_load_async_to_lds_b128(
          (__attribute__((address_space(1))) i32x4*)g,
          (__attribute__((address_space(3))) i32x4*)&As[r][ch << 3],
          0, 0);
    }
#else
    {
      uint4 av[2];
      #pragma unroll
      for (int t = 0; t < 2; ++t) {
        int idx = tid + t * 256;
        int r = idx >> 2, ch = idx & 3;
        av[t] = *(const uint4*)(A + (size_t)(m0 + r) * Kd + k0 + (ch << 3));
      }
      #pragma unroll
      for (int t = 0; t < 2; ++t) {
        int idx = tid + t * 256;
        int r = idx >> 2, ch = idx & 3;
        *(uint4*)&As[r][ch << 3] = av[t];
      }
    }
#endif

    // ---- Stage B tile transposed: global [n][k] -> LDS [k][n] -------------
    // Two-phase so the 8 global dword loads pipeline before the LDS scatter.
    {
      unsigned int bv[8];
      #pragma unroll
      for (int t = 0; t < 8; ++t) {
        int idx = tid + t * 256;
        int r = idx >> 4, d = idx & 15;
        bv[t] = *(const unsigned int*)(B + (size_t)(n0 + r) * Kd + k0 + (d << 1));
      }
      #pragma unroll
      for (int t = 0; t < 8; ++t) {
        int idx = tid + t * 256;
        int r = idx >> 4, d = idx & 15;
        Bs[(d << 1)][r]     = (unsigned short)(bv[t] & 0xFFFFu);
        Bs[(d << 1) + 1][r] = (unsigned short)(bv[t] >> 16);
      }
    }

#if __has_builtin(__builtin_amdgcn_global_load_async_to_lds_b128) && \
    __has_builtin(__builtin_amdgcn_s_wait_asynccnt)
    __builtin_amdgcn_s_wait_asynccnt(0);
#endif
    __syncthreads();

    // Prefetch next K tile into cache (-> global_prefetch_b8)
    if (kt + 1 < nk) {
      __builtin_prefetch(A + (size_t)(m0 + (tid >> 1)) * Kd + k0 + 32, 0, 1);
      __builtin_prefetch(B + (size_t)(n0 + (tid >> 1)) * Kd + k0 + 32, 0, 1);
    }

    // A fragment (16-bit 16x32 layout): lanes 0-15 hold K {0-7,16-23},
    // lanes 16-31 hold K {8-15,24-31} for row M = lane&15.
    FragAB fa[4], fb[2];
    #pragma unroll
    for (int i = 0; i < 4; ++i) {
      int mr = wm * 64 + i * 16 + mlane;
      fa[i].u4[0] = *(const uint4*)&As[mr][half << 3];
      fa[i].u4[1] = *(const uint4*)&As[mr][16 + (half << 3)];
    }
    // B fragment: lane -> n = lane&15, element s -> k in the same order as A
    #pragma unroll
    for (int j = 0; j < 2; ++j) {
      int nc = wn * 32 + j * 16 + mlane;
      #pragma unroll
      for (int s = 0; s < 16; ++s) {
        int kk = (s < 8) ? ((half << 3) + s) : (16 + (half << 3) + (s - 8));
        ((unsigned short*)&fb[j])[s] = Bs[kk][nc];
      }
    }
    #pragma unroll
    for (int i = 0; i < 4; ++i)
      #pragma unroll
      for (int j = 0; j < 2; ++j)
        acc[i][j] = __builtin_amdgcn_wmma_f32_16x16x32_bf16(
            false, fa[i].bf, false, fb[j].bf, (short)0, acc[i][j], false, false);
    __syncthreads();
  }

  // Epilogue: C/D 16x16 f32 layout: VGPR r -> M = r + 8*(lane>=16), N = lane&15
  const int gn_local = lane & 15;
  const int mhi = (lane >> 4) << 3;
  #pragma unroll
  for (int i = 0; i < 4; ++i) {
    #pragma unroll
    for (int j = 0; j < 2; ++j) {
      int gn  = n0 + wn * 32 + j * 16 + gn_local;
      int gmb = m0 + wm * 64 + i * 16 + mhi;
      #pragma unroll
      for (int r = 0; r < 8; ++r) {
        float v = acc[i][j][r] * alpha;
        size_t off = (size_t)blockIdx.z * (size_t)sC + (size_t)(gmb + r) * N + gn;
        if (outBf16) ((unsigned short*)Cg)[off] = f32_to_bf16_rne(v);
        else         ((float*)Cg)[off] = v;
      }
    }
  }
}

// ---------------------------------------------------------------------------
// 7) Row softmax (f32, in place in d_out) + post-softmax attention mask;
//    also emits a bf16 copy for the attn@V WMMA GEMM.
// ---------------------------------------------------------------------------
__global__ void __launch_bounds__(256)
softmax_mask_kernel(float* __restrict__ attn,
                    unsigned short* __restrict__ attnbf,
                    const float* __restrict__ mflag) {
  const int row = blockIdx.x;            // b*NP + q
  const int b = row >> 10;
  const int q = row & (NP - 1);
  const int tid = threadIdx.x;
  float* rp = attn + (size_t)row * NP;
  float4 x = ((const float4*)rp)[tid];

  __shared__ float red[256];
  red[tid] = fmaxf(fmaxf(x.x, x.y), fmaxf(x.z, x.w));
  __syncthreads();
  for (int s = 128; s > 0; s >>= 1) {
    if (tid < s) red[tid] = fmaxf(red[tid], red[tid + s]);
    __syncthreads();
  }
  const float mx = red[0];
  __syncthreads();

  float e0 = expf(x.x - mx), e1 = expf(x.y - mx);
  float e2 = expf(x.z - mx), e3 = expf(x.w - mx);
  red[tid] = e0 + e1 + e2 + e3;
  __syncthreads();
  for (int s = 128; s > 0; s >>= 1) {
    if (tid < s) red[tid] += red[tid + s];
    __syncthreads();
  }
  const float inv = 1.0f / red[0];

  const int kk0 = tid << 2;
  float a0 = e0 * inv, a1 = e1 * inv, a2 = e2 * inv, a3 = e3 * inv;
  const float is_mq = mflag[row];
  if (is_mq > 0.f) {
    const float* mf = mflag + ((size_t)b << 10) + kk0;
    a0 *= (1.f - mf[0]); a1 *= (1.f - mf[1]);
    a2 *= (1.f - mf[2]); a3 *= (1.f - mf[3]);
  } else {
    a0 = (kk0     == q) ? a0 : 0.f;
    a1 = (kk0 + 1 == q) ? a1 : 0.f;
    a2 = (kk0 + 2 == q) ? a2 : 0.f;
    a3 = (kk0 + 3 == q) ? a3 : 0.f;
  }
  ((float4*)rp)[tid] = make_float4(a0, a1, a2, a3);
  unsigned short* bp = attnbf + (size_t)row * NP + kk0;
  bp[0] = f32_to_bf16_rne(a0); bp[1] = f32_to_bf16_rne(a1);
  bp[2] = f32_to_bf16_rne(a2); bp[3] = f32_to_bf16_rne(a3);
}

// ---------------------------------------------------------------------------
// 8) Fold mixed patches back to pixels and composite with the mask
// ---------------------------------------------------------------------------
__global__ void composite_kernel(const float* __restrict__ img,
                                 const float* __restrict__ mask,
                                 const float* __restrict__ mixed,
                                 float* __restrict__ refined) {
  long long idx = (long long)blockIdx.x * blockDim.x + threadIdx.x;
  const long long total = (long long)BATCH * 3 * HW * HW;
  if (idx >= total) return;
  int x = (int)(idx & (HW - 1));
  int y = (int)((idx >> 9) & (HW - 1));
  int bo = (int)(idx >> 18);
  int c = bo % 3, b = bo / 3;
  int p = (y >> 4) * 32 + (x >> 4);
  int d = c * 256 + (y & 15) * 16 + (x & 15);
  float mv = mask[((size_t)b * HW + y) * HW + x];
  float mix = mixed[((size_t)b * NP + p) * PVD + d];
  refined[idx] = mix * mv + img[idx] * (1.f - mv);
}

// ---------------------------------------------------------------------------
// Host-side orchestration
// ---------------------------------------------------------------------------
static inline int nblk(long long n) { return (int)((n + 255) / 256); }

extern "C" void kernel_launch(void* const* d_in, const int* in_sizes, int n_in,
                              void* d_out, int out_size, void* d_ws, size_t ws_size,
                              hipStream_t stream) {
  (void)in_sizes; (void)n_in; (void)out_size; (void)ws_size;
  const float* image = (const float*)d_in[0];
  const float* mask  = (const float*)d_in[1];
  const float* stemw = (const float*)d_in[2];
  const float* feat  = (const float*)d_in[3];
  const float* wqs   = (const float*)d_in[4];
  const float* wks   = (const float*)d_in[5];
  const float* pos   = (const float*)d_in[6];

  float* out         = (float*)d_out;
  float* out_refined = out;                                       // B*3*512*512
  float* out_attn    = out + (size_t)BATCH * 3 * HW * HW;         // B*1024*1024
  float* out_coarse  = out_attn + (size_t)BATCH * NP * NP;        // B*3*512*512

  // Workspace layout (bytes)
  char* ws = (char*)d_ws;
  const size_t OFF_TOK    = 0;                                    // bf16 B*NP*PTD
  const size_t OFF_VT     = OFF_TOK    + (size_t)BATCH * NP  * PTD * 2;
  const size_t OFF_WQ     = OFF_VT     + (size_t)BATCH * PVD * NP  * 2;
  const size_t OFF_WK     = OFF_WQ     + (size_t)DQK * PTD * 2;
  const size_t OFF_Q      = OFF_WK     + (size_t)DQK * PTD * 2;
  const size_t OFF_K      = OFF_Q      + (size_t)BATCH * NP * DQK * 2;
  const size_t OFF_ATTNBF = OFF_K      + (size_t)BATCH * NP * DQK * 2;
  const size_t OFF_MIXED  = OFF_ATTNBF + (size_t)BATCH * NP * NP  * 2;
  const size_t OFF_MFLAG  = OFF_MIXED  + (size_t)BATCH * NP * PVD * 4;

  unsigned short* tok    = (unsigned short*)(ws + OFF_TOK);
  unsigned short* vt     = (unsigned short*)(ws + OFF_VT);
  unsigned short* wq_bf  = (unsigned short*)(ws + OFF_WQ);
  unsigned short* wk_bf  = (unsigned short*)(ws + OFF_WK);
  unsigned short* q_bf   = (unsigned short*)(ws + OFF_Q);
  unsigned short* k_bf   = (unsigned short*)(ws + OFF_K);
  unsigned short* attnbf = (unsigned short*)(ws + OFF_ATTNBF);
  float*          mixed  = (float*)(ws + OFF_MIXED);
  float*          mflag  = (float*)(ws + OFF_MFLAG);

  const dim3 blk(256);

  // 1) stem conv -> coarse output slot
  stem_conv_kernel<<<nblk((long long)BATCH * 3 * HW * HW), blk, 0, stream>>>(
      image, stemw, out_coarse);

  // 2) weights -> bf16
  f32_to_bf16_kernel<<<nblk((long long)DQK * PTD), blk, 0, stream>>>(wqs, wq_bf, (long long)DQK * PTD);
  f32_to_bf16_kernel<<<nblk((long long)DQK * PTD), blk, 0, stream>>>(wks, wk_bf, (long long)DQK * PTD);

  // 3) tokens / values / mask flags
  build_tokens_kernel<<<nblk((long long)BATCH * NP * PTD), blk, 0, stream>>>(
      out_coarse, feat, pos, tok);
  build_values_kernel<<<nblk((long long)BATCH * PVD * NP), blk, 0, stream>>>(image, vt);
  mask_flags_kernel<<<nblk(BATCH * NP), blk, 0, stream>>>(mask, mflag);

  const float qscale = 1.0f / sqrtf((float)DQK);

  // 4) Q = (tok @ Wq^T) * qscale   -> bf16   [M=1024, N=2048, K=896]
  gemm_bf16_wmma<<<dim3(DQK / 128, NP / 128, BATCH), blk, 0, stream>>>(
      tok, wq_bf, q_bf, NP, DQK, PTD,
      (long long)NP * PTD, 0LL, (long long)NP * DQK, qscale, 1);
  // 5) K = tok @ Wk^T              -> bf16
  gemm_bf16_wmma<<<dim3(DQK / 128, NP / 128, BATCH), blk, 0, stream>>>(
      tok, wk_bf, k_bf, NP, DQK, PTD,
      (long long)NP * PTD, 0LL, (long long)NP * DQK, 1.0f, 1);

  // 6) logits = Q @ K^T -> f32 directly into the attn output slot
  gemm_bf16_wmma<<<dim3(NP / 128, NP / 128, BATCH), blk, 0, stream>>>(
      q_bf, k_bf, out_attn, NP, NP, DQK,
      (long long)NP * DQK, (long long)NP * DQK, (long long)NP * NP, 1.0f, 0);

  // 7) softmax + attention mask (in place) + bf16 copy
  softmax_mask_kernel<<<BATCH * NP, blk, 0, stream>>>(out_attn, attnbf, mflag);

  // 8) mixed = attn @ V   [M=1024, N=768, K=1024]
  gemm_bf16_wmma<<<dim3(PVD / 128, NP / 128, BATCH), blk, 0, stream>>>(
      attnbf, vt, mixed, NP, PVD, NP,
      (long long)NP * NP, (long long)PVD * NP, (long long)NP * PVD, 1.0f, 0);

  // 9) fold + composite -> refined output slot
  composite_kernel<<<nblk((long long)BATCH * 3 * HW * HW), blk, 0, stream>>>(
      image, mask, mixed, out_refined);
}
